// MultiHeadAttentionBlock_54150947668773
// MI455X (gfx1250) — matrix operand order repaired and checked by reading, not compiled, not run on hardware
//
#include <hip/hip_runtime.h>

// ---------------------------------------------------------------------------
// MultiHeadAttentionBlock for MI455X (gfx1250)
// wave32 + v_wmma_f32_16x16x32_f16 + global_load_async_to_lds_b128 staging
// B=2, L=2048, D=1024, H=16, DK=64
// ---------------------------------------------------------------------------

#define BB 2
#define LL 2048
#define DD 1024
#define HH 16
#define DKK 64
#define NT (BB * LL) // 4096

typedef __attribute__((ext_vector_type(16))) _Float16 v16h;
typedef __attribute__((ext_vector_type(8)))  _Float16 v8h;
typedef __attribute__((ext_vector_type(8)))  float    v8f;

// ---- gfx1250 async global->LDS copy (ASYNCcnt path) -----------------------

#if defined(__has_builtin)
#  if __has_builtin(__builtin_amdgcn_global_load_async_to_lds_b128)
#    define HAVE_ASYNC_LDS_BUILTIN 1
#  endif
#  if __has_builtin(__builtin_amdgcn_s_wait_asynccnt)
#    define HAVE_ASYNC_WAIT_BUILTIN 1
#  endif
#endif

// Parameter type per hipcc diagnostic: 'int __attribute__((vector_size(16))) AS*'
typedef int vint4 __attribute__((vector_size(4 * sizeof(int))));
typedef __attribute__((address_space(1))) vint4 g_vint4;   // global (__device__)
typedef __attribute__((address_space(3))) vint4 l_vint4;   // LDS (__shared__)

__device__ __forceinline__ void async_copy16(void* lds_dst, const void* gsrc) {
#if defined(HAVE_ASYNC_LDS_BUILTIN)
    __builtin_amdgcn_global_load_async_to_lds_b128(
        (g_vint4*)gsrc, (l_vint4*)lds_dst, /*offset=*/0, /*cpol=*/0);
#else
    unsigned lds_off = (unsigned)(unsigned long long)lds_dst;
    asm volatile("global_load_async_to_lds_b128 %0, %1, off"
                 :: "v"(lds_off), "v"(gsrc) : "memory");
#endif
}

template <int N>
__device__ __forceinline__ void wait_asynccnt() {
#if defined(HAVE_ASYNC_WAIT_BUILTIN)
    __builtin_amdgcn_s_wait_asynccnt(N);
#else
    asm volatile("s_wait_asynccnt %0" :: "i"(N) : "memory");
#endif
}

// ---- WMMA helpers ---------------------------------------------------------

__device__ __forceinline__ v8f wmma32(v16h a, v16h b, v8f c) {
    // D = A(16x32 f16) * B(32x16 f16) + C(16x16 f32)
    return __builtin_amdgcn_wmma_f32_16x16x32_f16(
        /*neg_a=*/false, a, /*neg_b=*/false, b,
        /*c_mod=*/(short)0, c, /*reuse_a=*/false, /*reuse_b=*/false);
}

// A-matrix fragment (16x32, f16). Lane l: row = row0 + (l&15),
// elements 0..7  = K (kb..kb+7), elements 8..15 = K (kb+16..kb+23), kb=8*(l>>4).
__device__ __forceinline__ v16h load_a_frag(const _Float16* base, int stride,
                                            int row0, int k0) {
    int lane = threadIdx.x & 31;
    int m  = row0 + (lane & 15);
    int kb = k0 + ((lane >> 4) << 3);
    const v8h* p0 = (const v8h*)(base + (size_t)m * stride + kb);
    const v8h* p1 = (const v8h*)(base + (size_t)m * stride + kb + 16);
    v8h lo = *p0, hi = *p1;
    v16h r;
#pragma unroll
    for (int i = 0; i < 8; ++i) { r[i] = lo[i]; r[i + 8] = hi[i]; }
    return r;
}

// B-matrix fragment (32x16, f16). Lane l: column n = n0 + (l&15),
// elements 0..7 = K (kk..kk+7), elements 8..15 = K (kk+16..kk+23), kk = k0 + 8*(l>>4).
// `colbase` stores B column-major: B[k][n] at colbase[n*stride + k].
__device__ __forceinline__ v16h load_b_frag(const _Float16* colbase, int stride,
                                            int n0, int k0) {
    int lane = threadIdx.x & 31;
    int n  = n0 + (lane & 15);
    int kk = k0 + ((lane >> 4) << 3);
    const v8h* p = (const v8h*)(colbase + (size_t)n * stride + kk);
    v8h lo = p[0], hi = p[2];
    v16h r;
#pragma unroll
    for (int i = 0; i < 8; ++i) { r[i] = lo[i]; r[i + 8] = hi[i]; }
    return r;
}

// ------------------------------ prep kernels -------------------------------

__global__ void cvt_f32_to_f16(const float* __restrict__ s,
                               _Float16* __restrict__ d, int n) {
    int i = blockIdx.x * blockDim.x + threadIdx.x;
    if (i < n) d[i] = (_Float16)s[i];
}

// W[rows][cols] f32 (din x dout, row-major)  ->  WT[cols][rows] f16
__global__ void transpose_cvt(const float* __restrict__ W,
                              _Float16* __restrict__ WT, int rows, int cols) {
    int i = blockIdx.x * blockDim.x + threadIdx.x;
    if (i >= rows * cols) return;
    int n = i / rows;
    int k = i % rows;
    WT[(size_t)n * rows + k] = (_Float16)W[(size_t)k * cols + n];
}

// ------------------------------ GEMM kernel --------------------------------
// C[4096x1024] = X[4096x1024](f16) @ W(via WT [N][K] f16) + bias(f32)
// Shared B-tile (64 cols x 32 k) is async-DMA'd into LDS, double-buffered.
// MODE 0: out f16 -> [B,H,L,DK]   (Q, K projections)
// MODE 1: out f16 -> [B,H,DK,L]   (V projection, pre-transposed)
// MODE 2: out f32 -> [M][N]       (final output projection)
template <int MODE>
__global__ void gemm_kernel(const _Float16* __restrict__ X,
                            const _Float16* __restrict__ WT,
                            const float* __restrict__ bias,
                            void* __restrict__ out) {
    __shared__ alignas(16) _Float16 btile[2][64 * 32];   // 2 x 4KB
    const int K = DD;
    int wave = threadIdx.x >> 5;
    int lane = threadIdx.x & 31;
    int m0 = blockIdx.x * 64 + wave * 16;
    int n0 = blockIdx.y * 64;

    // stage WT[n0..n0+63][kk..kk+31] into btile[s] (each thread: 2 x 16B)
    auto stage = [&](int kk, int s) {
#pragma unroll
        for (int q = 0; q < 2; ++q) {
            int c   = threadIdx.x * 2 + q;   // 0..255 16B chunks
            int col = c >> 2;                // 0..63
            int ko  = (c & 3) * 8;           // half offset in k
            async_copy16(&btile[s][col * 32 + ko],
                         WT + (size_t)(n0 + col) * K + kk + ko);
        }
    };

    v8f acc[4];
#pragma unroll
    for (int j = 0; j < 4; ++j) {
        float bv = bias[n0 + 16 * j + (lane & 15)];
#pragma unroll
        for (int r = 0; r < 8; ++r) acc[j][r] = bv;
    }

    stage(0, 0);
    int s = 0;
    for (int kk = 0; kk < K; kk += 32, s ^= 1) {
        if (kk + 32 < K) {
            stage(kk + 32, s ^ 1);
            wait_asynccnt<2>();      // current stage landed (next stays inflight)
        } else {
            wait_asynccnt<0>();
        }
        __syncthreads();             // all waves see btile[s]

        if ((kk & 127) == 0 && kk + 128 < K)
            __builtin_prefetch(X + (size_t)(m0 + (lane & 15)) * K + kk + 128, 0, 1);
        v16h a = load_a_frag(X, K, m0, kk);
#pragma unroll
        for (int j = 0; j < 4; ++j)
            acc[j] = wmma32(a, load_b_frag(&btile[s][0], 32, 16 * j, 0), acc[j]);

        __syncthreads();             // done reading btile[s] before re-staging it
    }

    int mrow = m0 + ((lane >> 4) << 3);
#pragma unroll
    for (int j = 0; j < 4; ++j) {
        int n = n0 + 16 * j + (lane & 15);
#pragma unroll
        for (int r = 0; r < 8; ++r) {
            int m = mrow + r;
            float v = acc[j][r];
            if (MODE == 2) {
                ((float*)out)[(size_t)m * DD + n] = v;
            } else {
                int b = m >> 11, l = m & (LL - 1);
                int h = n >> 6, dk = n & (DKK - 1);
                _Float16 hv = (_Float16)v;
                if (MODE == 0)
                    ((_Float16*)out)[(((size_t)(b * HH + h)) * LL + l) * DKK + dk] = hv;
                else
                    ((_Float16*)out)[(((size_t)(b * HH + h)) * DKK + dk) * LL + l] = hv;
            }
        }
    }
}

// --------------------------- flash attention -------------------------------
// grid: (L/64, B*H), block 128 (4 waves). Wave w handles q rows [q0, q0+16).
// K-tile (32 keys x 64 dk) and V-tile (64 dk x 32 keys, from pre-transposed V)
// are shared by all 4 waves -> async-DMA into LDS, double-buffered.
__global__ void attn_kernel(const _Float16* __restrict__ Q,
                            const _Float16* __restrict__ Km,
                            const _Float16* __restrict__ VT,
                            const int* __restrict__ mask,
                            _Float16* __restrict__ heads) {
    __shared__ alignas(16) _Float16 ktile[2][32 * 64];   // 2 x 4KB
    __shared__ alignas(16) _Float16 vtile[2][64 * 32];   // 2 x 4KB
    __shared__ alignas(16) _Float16 ptile[4][16 * 32];   // per-wave P
    int wave = threadIdx.x >> 5;
    int lane = threadIdx.x & 31;
    int bh = blockIdx.y;
    int b = bh >> 4, h = bh & 15;
    int q0 = blockIdx.x * 64 + wave * 16;

    const _Float16* Qb = Q  + (size_t)bh * LL * DKK;
    const _Float16* Kb = Km + (size_t)bh * LL * DKK;
    const _Float16* Vb = VT + (size_t)bh * DKK * LL;     // [dk][l]
    const int* maskb   = mask + (size_t)b * LL * LL;

    const float scale = 0.125f;                          // 1/sqrt(64)
    int rbase = (lane >> 4) << 3;
    int colk  = lane & 15;

    // stage K rows [kb..kb+31] (rows of 64 halves) and V rows [dk][kb..kb+31]
    auto stage = [&](int kb, int s) {
#pragma unroll
        for (int q = 0; q < 2; ++q) {
            int c   = threadIdx.x * 2 + q;   // 0..255
            int key = c >> 3;                // 0..31
            int ko  = (c & 7) * 8;           // 0..56 halves
            async_copy16(&ktile[s][key * 64 + ko],
                         Kb + (size_t)(kb + key) * DKK + ko);
        }
#pragma unroll
        for (int q = 0; q < 2; ++q) {
            int c  = threadIdx.x * 2 + q;    // 0..255
            int dk = c >> 2;                 // 0..63
            int ko = (c & 3) * 8;            // 0..24 halves
            async_copy16(&vtile[s][dk * 32 + ko],
                         Vb + (size_t)dk * LL + kb + ko);
        }
    };

    v16h aq0 = load_a_frag(Qb, DKK, q0, 0);
    v16h aq1 = load_a_frag(Qb, DKK, q0, 32);

    v8f o[4];
#pragma unroll
    for (int j = 0; j < 4; ++j)
#pragma unroll
        for (int r = 0; r < 8; ++r) o[j][r] = 0.f;
    float mrow[8], lrow[8];
#pragma unroll
    for (int r = 0; r < 8; ++r) { mrow[r] = -3.0e38f; lrow[r] = 0.f; }

    stage(0, 0);
    int s = 0;
    for (int kb = 0; kb < LL; kb += 32, s ^= 1) {
        if (kb + 32 < LL) {
            stage(kb + 32, s ^ 1);
            wait_asynccnt<4>();
        } else {
            wait_asynccnt<0>();
        }
        __syncthreads();

        const _Float16* kt = &ktile[s][0];
        const _Float16* vt = &vtile[s][0];

        // S = Q * K^T for 32 keys (two 16-key accumulators), from LDS tiles
        v8f s0, s1;
#pragma unroll
        for (int r = 0; r < 8; ++r) { s0[r] = 0.f; s1[r] = 0.f; }
        s0 = wmma32(aq0, load_b_frag(kt, DKK, 0, 0),  s0);
        s0 = wmma32(aq1, load_b_frag(kt, DKK, 0, 32), s0);
        s1 = wmma32(aq0, load_b_frag(kt, DKK, 16, 0),  s1);
        s1 = wmma32(aq1, load_b_frag(kt, DKK, 16, 32), s1);

        // scale + mask + online softmax (per accumulator slot / q row)
#pragma unroll
        for (int r = 0; r < 8; ++r) {
            int qrow = q0 + rbase + r;
            float f0 = s0[r] * scale;
            float f1 = s1[r] * scale;
            if (maskb[(size_t)qrow * LL + kb + colk] == 0)      f0 = -1.0e30f;
            if (maskb[(size_t)qrow * LL + kb + 16 + colk] == 0) f1 = -1.0e30f;
            float mx = fmaxf(f0, f1);
#pragma unroll
            for (int d = 1; d < 16; d <<= 1) mx = fmaxf(mx, __shfl_xor(mx, d, 32));
            float mn = fmaxf(mrow[r], mx);
            float corr = __expf(mrow[r] - mn);
            float p0 = __expf(f0 - mn);
            float p1 = __expf(f1 - mn);
            float rs = p0 + p1;
#pragma unroll
            for (int d = 1; d < 16; d <<= 1) rs += __shfl_xor(rs, d, 32);
            lrow[r] = lrow[r] * corr + rs;
            mrow[r] = mn;
#pragma unroll
            for (int j = 0; j < 4; ++j) o[j][r] *= corr;
            ptile[wave][(rbase + r) * 32 + colk]      = (_Float16)p0;
            ptile[wave][(rbase + r) * 32 + 16 + colk] = (_Float16)p1;
        }

        // O += P(16x32) * V(32xDK) from LDS V-tile
        v16h ap = load_a_frag(&ptile[wave][0], 32, 0, 0);
#pragma unroll
        for (int j = 0; j < 4; ++j)
            o[j] = wmma32(ap, load_b_frag(vt, 32, 16 * j, 0), o[j]);

        __syncthreads();             // tiles consumed before re-staging
    }

    // normalize and store heads [B,L,H,DK] (f16)
#pragma unroll
    for (int j = 0; j < 4; ++j) {
        int dk = 16 * j + colk;
#pragma unroll
        for (int r = 0; r < 8; ++r) {
            int qrow = q0 + rbase + r;
            float v = o[j][r] / lrow[r];
            heads[(((size_t)(b * LL + qrow)) * HH + h) * DKK + dk] = (_Float16)v;
        }
    }
}

// --------------------------- inter-head attention --------------------------
// One wave per (b,l): heads[16][64] -> Qh/Kh/Vh -> 16x16 softmax -> mixed.
__global__ void interhead_kernel(const _Float16* __restrict__ heads,
                                 const _Float16* __restrict__ whqT,
                                 const _Float16* __restrict__ whkT,
                                 const _Float16* __restrict__ whvT,
                                 const float* __restrict__ bhq,
                                 const float* __restrict__ bhk,
                                 const float* __restrict__ bhv,
                                 _Float16* __restrict__ mixed) {
    __shared__ alignas(16) _Float16 qh[4][16 * 64];
    __shared__ alignas(16) _Float16 kh[4][16 * 64];
    __shared__ alignas(16) _Float16 vhT[4][64 * 32];
    __shared__ alignas(16) _Float16 pt[4][16 * 32];

    int wave = threadIdx.x >> 5;
    int lane = threadIdx.x & 31;
    int bl = blockIdx.x * 4 + wave;
    int rbase = (lane >> 4) << 3;
    int coln  = lane & 15;

    const _Float16* Hm = heads + (size_t)bl * DD;   // [16][64]
    v16h a0 = load_a_frag(Hm, DKK, 0, 0);
    v16h a1 = load_a_frag(Hm, DKK, 0, 32);

    // zero-pad vhT key columns 16..31
    for (int i = lane; i < 64 * 16; i += 32) {
        int dk = i >> 4, g = 16 + (i & 15);
        vhT[wave][dk * 32 + g] = (_Float16)0.f;
    }

    // Qh / Kh -> LDS [h][dk];  Vh -> LDS transposed [dk][g]
#pragma unroll
    for (int p = 0; p < 3; ++p) {
        const _Float16* WT = (p == 0) ? whqT : (p == 1) ? whkT : whvT;
        const float* bia   = (p == 0) ? bhq  : (p == 1) ? bhk  : bhv;
#pragma unroll
        for (int j = 0; j < 4; ++j) {
            v8f acc;
            float bv = bia[16 * j + coln];
#pragma unroll
            for (int r = 0; r < 8; ++r) acc[r] = bv;
            acc = wmma32(a0, load_b_frag(WT, DKK, 16 * j, 0), acc);
            acc = wmma32(a1, load_b_frag(WT, DKK, 16 * j, 32), acc);
#pragma unroll
            for (int r = 0; r < 8; ++r) {
                if (p == 0)      qh[wave][(rbase + r) * 64 + 16 * j + coln] = (_Float16)acc[r];
                else if (p == 1) kh[wave][(rbase + r) * 64 + 16 * j + coln] = (_Float16)acc[r];
                else             vhT[wave][(16 * j + coln) * 32 + rbase + r] = (_Float16)acc[r];
            }
        }
    }

    // inter = (Qh @ Kh^T) * scale, softmax over 16 head columns
    v16h aq0 = load_a_frag(&qh[wave][0], 64, 0, 0);
    v16h aq1 = load_a_frag(&qh[wave][0], 64, 0, 32);
    v8f sc;
#pragma unroll
    for (int r = 0; r < 8; ++r) sc[r] = 0.f;
    sc = wmma32(aq0, load_b_frag(&kh[wave][0], 64, 0, 0), sc);
    sc = wmma32(aq1, load_b_frag(&kh[wave][0], 64, 0, 32), sc);

#pragma unroll
    for (int r = 0; r < 8; ++r) {
        float v = sc[r] * 0.125f;
        float mx = v;
#pragma unroll
        for (int d = 1; d < 16; d <<= 1) mx = fmaxf(mx, __shfl_xor(mx, d, 32));
        float pv = __expf(v - mx);
        float sum = pv;
#pragma unroll
        for (int d = 1; d < 16; d <<= 1) sum += __shfl_xor(sum, d, 32);
        pv /= sum;
        pt[wave][(rbase + r) * 32 + coln]      = (_Float16)pv;
        pt[wave][(rbase + r) * 32 + 16 + coln] = (_Float16)0.f;  // K pad
    }

    // mixed = P(16x16, zero-padded to K=32) @ Vh(16x64)
    v16h ap = load_a_frag(&pt[wave][0], 32, 0, 0);
#pragma unroll
    for (int j = 0; j < 4; ++j) {
        v8f m;
#pragma unroll
        for (int r = 0; r < 8; ++r) m[r] = 0.f;
        m = wmma32(ap, load_b_frag(&vhT[wave][0], 32, 16 * j, 0), m);
#pragma unroll
        for (int r = 0; r < 8; ++r)
            mixed[(size_t)bl * DD + (rbase + r) * DKK + 16 * j + coln] = (_Float16)m[r];
    }
}

// ------------------------------- launcher ----------------------------------

extern "C" void kernel_launch(void* const* d_in, const int* in_sizes, int n_in,
                              void* d_out, int out_size, void* d_ws, size_t ws_size,
                              hipStream_t stream) {
    (void)in_sizes; (void)n_in; (void)out_size; (void)ws_size;

    const float* q_in = (const float*)d_in[0];
    const float* k_in = (const float*)d_in[1];
    const float* v_in = (const float*)d_in[2];
    const int*   mask = (const int*)  d_in[3];
    const float* wq = (const float*)d_in[4];  const float* bq = (const float*)d_in[5];
    const float* wk = (const float*)d_in[6];  const float* bk = (const float*)d_in[7];
    const float* wv = (const float*)d_in[8];  const float* bv = (const float*)d_in[9];
    const float* wo = (const float*)d_in[10]; const float* bo = (const float*)d_in[11];
    const float* whq = (const float*)d_in[12]; const float* bhq = (const float*)d_in[13];
    const float* whk = (const float*)d_in[14]; const float* bhk = (const float*)d_in[15];
    const float* whv = (const float*)d_in[16]; const float* bhv = (const float*)d_in[17];
    // d_in[18]/d_in[19] (who/bho) are dead in the reference forward.

    char* ws = (char*)d_ws;
    size_t off = 0;
    auto alloc = [&](size_t bytes) -> void* {
        void* p = ws + off;
        off = (off + bytes + 255) & ~(size_t)255;
        return p;
    };

    const size_t NE = (size_t)NT * DD;             // 4M elements
    _Float16* Xq16 = (_Float16*)alloc(NE * 2);
    _Float16* Xk16 = (_Float16*)alloc(NE * 2);
    _Float16* Xv16 = (_Float16*)alloc(NE * 2);
    _Float16* WqT  = (_Float16*)alloc((size_t)DD * DD * 2);
    _Float16* WkT  = (_Float16*)alloc((size_t)DD * DD * 2);
    _Float16* WvT  = (_Float16*)alloc((size_t)DD * DD * 2);
    _Float16* WoT  = (_Float16*)alloc((size_t)DD * DD * 2);
    _Float16* WhqT = (_Float16*)alloc((size_t)DKK * DKK * 2);
    _Float16* WhkT = (_Float16*)alloc((size_t)DKK * DKK * 2);
    _Float16* WhvT = (_Float16*)alloc((size_t)DKK * DKK * 2);
    _Float16* Q16  = (_Float16*)alloc(NE * 2);     // [B,H,L,DK]
    _Float16* K16  = (_Float16*)alloc(NE * 2);     // [B,H,L,DK]
    _Float16* VT16 = (_Float16*)alloc(NE * 2);     // [B,H,DK,L]
    _Float16* Hd16 = (_Float16*)alloc(NE * 2);     // heads [B,L,H,DK]
    _Float16* Mx16 = (_Float16*)alloc(NE * 2);     // mixed [B,L,D]

    // 1) convert inputs to f16
    cvt_f32_to_f16<<<(int)(NE / 256), 256, 0, stream>>>(q_in, Xq16, (int)NE);
    cvt_f32_to_f16<<<(int)(NE / 256), 256, 0, stream>>>(k_in, Xk16, (int)NE);
    cvt_f32_to_f16<<<(int)(NE / 256), 256, 0, stream>>>(v_in, Xv16, (int)NE);

    // 2) transpose+convert weights ([K][N] f32 -> [N][K] f16)
    int nw = DD * DD;
    transpose_cvt<<<(nw + 255) / 256, 256, 0, stream>>>(wq, WqT, DD, DD);
    transpose_cvt<<<(nw + 255) / 256, 256, 0, stream>>>(wk, WkT, DD, DD);
    transpose_cvt<<<(nw + 255) / 256, 256, 0, stream>>>(wv, WvT, DD, DD);
    transpose_cvt<<<(nw + 255) / 256, 256, 0, stream>>>(wo, WoT, DD, DD);
    int nh = DKK * DKK;
    transpose_cvt<<<(nh + 255) / 256, 256, 0, stream>>>(whq, WhqT, DKK, DKK);
    transpose_cvt<<<(nh + 255) / 256, 256, 0, stream>>>(whk, WhkT, DKK, DKK);
    transpose_cvt<<<(nh + 255) / 256, 256, 0, stream>>>(whv, WhvT, DKK, DKK);

    // 3) Q/K/V projections (V written pre-transposed for flash-attention B-frags)
    dim3 ggrid(NT / 64, DD / 64);
    gemm_kernel<0><<<ggrid, 128, 0, stream>>>(Xq16, WqT, bq, (void*)Q16);
    gemm_kernel<0><<<ggrid, 128, 0, stream>>>(Xk16, WkT, bk, (void*)K16);
    gemm_kernel<1><<<ggrid, 128, 0, stream>>>(Xv16, WvT, bv, (void*)VT16);

    // 4) flash attention per (b,h)
    attn_kernel<<<dim3(LL / 64, BB * HH), 128, 0, stream>>>(Q16, K16, VT16, mask, Hd16);

    // 5) inter-head attention per (b,l)
    interhead_kernel<<<NT / 4, 128, 0, stream>>>(Hd16, WhqT, WhkT, WhvT,
                                                 bhq, bhk, bhv, Mx16);

    // 6) output projection -> f32 d_out [B,L,D]
    gemm_kernel<2><<<ggrid, 128, 0, stream>>>(Mx16, WoT, bo, d_out);
}
